// H2VQFlow_28741921145216
// MI455X (gfx1250) — matrix operand order repaired and checked by reading, not compiled, not run on hardware
//
#include <hip/hip_runtime.h>
#include <stdint.h>

typedef __attribute__((ext_vector_type(16))) _Float16 v16h;
typedef __attribute__((ext_vector_type(8)))  float    v8f;

#define N_ROWS   32768                      // B*T = 16*2048
#define DIM      64
#define KCB      8192
#define ROW_TILE 16
#define CB_TILE  16
#define N_TILES  (KCB / CB_TILE)            // 512
#define WAVES    8
#define M_TILES  4                          // row tiles per block (B-fragment reuse)
#define ROWS_PER_BLOCK (ROW_TILE * M_TILES) // 64
#define XQ_ELEMS (N_ROWS * DIM)             // 2097152 floats
#define WS_CB_DWORDS (N_TILES * 2 * 32 * 8) // 262144 dwords = 1 MB packed f16 codebook

static __device__ inline uint32_t pack_f16x2(float a, float b) {
    _Float16 ha = (_Float16)a, hb = (_Float16)b;
    unsigned short ua, ub;
    __builtin_memcpy(&ua, &ha, 2);
    __builtin_memcpy(&ub, &hb, 2);
    return (uint32_t)ua | ((uint32_t)ub << 16);
}

// ---------------------------------------------------------------------------
// Kernel 1: repack fp32 codebook [K, 64] into f16 WMMA B-fragment order.
// ws_cb dword index = t*512 + c*256 + lane*8 + j
//   entry = t*16 + (lane&15), half = lane>>4, d = c*32 + half*16 + 2*j (+slot)
// ---------------------------------------------------------------------------
__global__ __launch_bounds__(256) void pack_cb_kernel(
    const float* __restrict__ cb, uint32_t* __restrict__ ws_cb)
{
    int tid = blockIdx.x * blockDim.x + threadIdx.x;  // 0 .. WS_CB_DWORDS-1
    int j    = tid & 7;
    int lane = (tid >> 3) & 31;
    int c    = (tid >> 8) & 1;
    int t    = tid >> 9;
    int entry = t * CB_TILE + (lane & 15);
    int half  = lane >> 4;
    int d     = c * 32 + half * 16 + 2 * j;
    const float* row = cb + (size_t)entry * DIM + d;
    ws_cb[tid] = pack_f16x2(row[0], row[1]);
}

// ---------------------------------------------------------------------------
// Kernel 2: ne2h[k] = -||codebook[k]||^2 / 2 (fp32) — used to seed WMMA's C
// matrix so acc = x.e - e^2/2 comes out of the MMA directly (argmax form).
// ---------------------------------------------------------------------------
__global__ __launch_bounds__(256) void ne2h_kernel(
    const float* __restrict__ cb, float* __restrict__ ne2h)
{
    int k = blockIdx.x * blockDim.x + threadIdx.x;    // 0 .. KCB-1
    const float4* p = (const float4*)(cb + (size_t)k * DIM);
    float s = 0.0f;
#pragma unroll
    for (int i = 0; i < DIM / 4; ++i) {
        float4 v = p[i];
        s = fmaf(v.x, v.x, s);
        s = fmaf(v.y, v.y, s);
        s = fmaf(v.z, v.z, s);
        s = fmaf(v.w, v.w, s);
    }
    ne2h[k] = -0.5f * s;
}

// ---------------------------------------------------------------------------
// Kernel 3: fused GEMM + argmax + gather.
// One block = 4 row tiles (64 x-rows); 8 waves stride the 512 codebook tiles.
// B fragments loaded once per block-iteration feed 4 WMMA chains; the C
// matrix is seeded with -e^2/2 so no post-MMA FMA is needed.
//   argmin_k ||x - e_k||^2  ==  argmax_k (x.e_k - ||e_k||^2/2)
// ---------------------------------------------------------------------------
__global__ __launch_bounds__(256) void vq_argmin_kernel(
    const float* __restrict__ x,
    const float* __restrict__ cb,
    const uint32_t* __restrict__ ws_cb,
    const float* __restrict__ ws_ne2h,
    float* __restrict__ out_xq,
    int* __restrict__ out_idx)
{
    __shared__ float s_rval[M_TILES][ROW_TILE][WAVES];
    __shared__ int   s_ridx[M_TILES][ROW_TILE][WAVES];
    __shared__ int   s_win[ROWS_PER_BLOCK];

    const int block_row0 = blockIdx.x * ROWS_PER_BLOCK;
    const int wave = threadIdx.x >> 5;
    const int lane = threadIdx.x & 31;
    const int row  = lane & 15;   // doubles as C-matrix column id
    const int half = lane >> 4;

    // --- Build A fragments for the 4 row tiles (each 16x64 as two 16x32 f16) ---
    // A 16x32 layout (matches packed B): lanes 0-15 -> M=lane, elem e: K=half*16+e;
    // lane-half picks the +16 K offset.
    v16h a[M_TILES][2];
#pragma unroll
    for (int mt = 0; mt < M_TILES; ++mt) {
        const float* xrow =
            x + (size_t)(block_row0 + mt * ROW_TILE + row) * DIM;
#pragma unroll
        for (int c = 0; c < 2; ++c) {
            const int b0 = c * 32 + half * 16;
            const int b1 = c * 32 + half * 16 + 8;
#pragma unroll
            for (int i = 0; i < 8; ++i) a[mt][c][i]     = (_Float16)xrow[b0 + i];
#pragma unroll
            for (int i = 0; i < 8; ++i) a[mt][c][8 + i] = (_Float16)xrow[b1 + i];
        }
    }

    float maxv[M_TILES][8];
    int   maxi[M_TILES][8];
#pragma unroll
    for (int mt = 0; mt < M_TILES; ++mt)
#pragma unroll
        for (int r = 0; r < 8; ++r) { maxv[mt][r] = -3.4e38f; maxi[mt][r] = 0; }

    // --- Sweep codebook tiles: load B once, seed C, 4x(2 WMMAs + argmax) ---
#pragma unroll 1
    for (int i = 0; i < N_TILES / WAVES; ++i) {
        const int t = wave + i * WAVES;
        const uint32_t* bp = ws_cb + (size_t)t * 512 + lane * 8;
        v16h b0 = *(const v16h*)(bp);          // chunk 0: d = 0..31
        v16h b1 = *(const v16h*)(bp + 256);    // chunk 1: d = 32..63
        // Unconditional prefetch of this wave's next tile (stays inside ws).
        __builtin_prefetch(bp + 8 * 512, 0, 1);
        const float ne2  = ws_ne2h[t * CB_TILE + row];
        const int  entry = t * CB_TILE + row;

        // C seed: each lane's column value broadcast to all 8 row slots;
        // shared by all 4 row-tile accumulator chains.
        v8f cinit;
#pragma unroll
        for (int r = 0; r < 8; ++r) cinit[r] = ne2;

#pragma unroll
        for (int mt = 0; mt < M_TILES; ++mt) {
            v8f acc = __builtin_amdgcn_wmma_f32_16x16x32_f16(
                          false, a[mt][0], false, b0, (short)0, cinit, false, false);
            acc = __builtin_amdgcn_wmma_f32_16x16x32_f16(
                          false, a[mt][1], false, b1, (short)0, acc, false, false);
#pragma unroll
            for (int r = 0; r < 8; ++r) {
                // C layout: lane l holds col N=l&15, acc[r] is row M=r+8*(l>>4)
                if (acc[r] > maxv[mt][r]) { maxv[mt][r] = acc[r]; maxi[mt][r] = entry; }
            }
        }
    }

    // --- In-wave reduction across the 16 columns (xor masks stay in-half) ---
#pragma unroll
    for (int mt = 0; mt < M_TILES; ++mt) {
#pragma unroll
        for (int r = 0; r < 8; ++r) {
            float v  = maxv[mt][r];
            int   ix = maxi[mt][r];
#pragma unroll
            for (int off = 1; off < 16; off <<= 1) {
                float ov = __shfl_xor(v, off, 32);
                int   oi = __shfl_xor(ix, off, 32);
                if (ov > v || (ov == v && oi < ix)) { v = ov; ix = oi; }
            }
            if (row == 0) {                    // lanes 0 and 16 publish
                const int m = r + 8 * half;    // row within the tile
                s_rval[mt][m][wave] = v;
                s_ridx[mt][m][wave] = ix;
            }
        }
    }
    __syncthreads();

    // --- Cross-wave reduction: one thread per block row (64 rows) ---
    if (threadIdx.x < ROWS_PER_BLOCK) {
        const int mt = threadIdx.x >> 4;
        const int m  = threadIdx.x & 15;
        float best = -3.4e38f;
        int   bi   = 0x7fffffff;
#pragma unroll
        for (int w = 0; w < WAVES; ++w) {
            float v = s_rval[mt][m][w];
            int   i = s_ridx[mt][m][w];
            // jnp.argmin tie-break: lowest index wins
            if (v > best || (v == best && i < bi)) { best = v; bi = i; }
        }
        s_win[threadIdx.x] = bi;
        out_idx[block_row0 + threadIdx.x] = bi;
    }
    __syncthreads();

    // --- Gather winning codebook rows (fp32, exact) -> x_q_st ---
    // 64 rows * 64 elems = 4096 floats; 256 threads * 4 x float4 each.
    const int m_all = threadIdx.x >> 2;            // block row 0..63
    const int d0    = (threadIdx.x & 3) * 16;      // 16-float slice
    const float4* src = (const float4*)(cb + (size_t)s_win[m_all] * DIM + d0);
    float4* dst = (float4*)(out_xq + (size_t)(block_row0 + m_all) * DIM + d0);
#pragma unroll
    for (int q = 0; q < 4; ++q) dst[q] = src[q];
}

extern "C" void kernel_launch(void* const* d_in, const int* in_sizes, int n_in,
                              void* d_out, int out_size, void* d_ws, size_t ws_size,
                              hipStream_t stream) {
    (void)in_sizes; (void)n_in; (void)out_size; (void)ws_size;
    const float* x  = (const float*)d_in[0];   // [16, 2048, 64] fp32
    const float* cb = (const float*)d_in[1];   // [8192, 64] fp32

    uint32_t* ws_cb   = (uint32_t*)d_ws;                                  // 1 MB
    float*    ws_ne2h = (float*)((char*)d_ws + (size_t)WS_CB_DWORDS * 4); // 32 KB

    float* out_xq  = (float*)d_out;               // x_q_st: 2097152 fp32
    int*   out_idx = (int*)d_out + XQ_ELEMS;      // indices: 32768 int32 (raw bits)

    pack_cb_kernel<<<WS_CB_DWORDS / 256, 256, 0, stream>>>(cb, ws_cb);
    ne2h_kernel<<<KCB / 256, 256, 0, stream>>>(cb, ws_ne2h);
    vq_argmin_kernel<<<N_ROWS / ROWS_PER_BLOCK, 256, 0, stream>>>(
        x, cb, ws_cb, ws_ne2h, out_xq, out_idx);
}
